// TDAModel_12524124635956
// MI455X (gfx1250) — matrix operand order repaired and checked by reading, not compiled, not run on hardware
//
#include <hip/hip_runtime.h>
#include <math.h>

// ---------------------------------------------------------------------------
// CDNA5 / gfx1250 implementation of the EDVR-style PCD alignment reference.
// Every conv + dcn einsum runs through v_wmma_f32_16x16x32_bf16.
// All feature tensors are NHWC (C=16; bf16 GEMM operands, fp32 masters for
// bilinear-sampled tensors / dcn offsets). With the ISA 16-bit A-fragment
// layout each half-fragment is 8 consecutive channels of one tap -> one
// global_load_b128 per chunk; weights are packed fragment-major -> 2x b128
// per K-step. k_conv is templated on <CINSHIFT,KSTEPS> so the K loop fully
// unrolls: tap geometry (div-by-3), the zero pad tap, and the concat A/B
// tensor select all constant-fold. All spatial sizes (160^2, 80^2, 40^2) are
// multiples of 16, so M tiles are always full (no tail predication).
// Workload is memory bound (~100 GFLOP vs ~1.5 GB @ 23.3 TB/s): bf16 + wide
// loads are the win; WMMA N=16 maps exactly onto nf=16.
// ---------------------------------------------------------------------------

typedef __bf16 bf16_t;
typedef __attribute__((ext_vector_type(16))) __bf16 v16bf;
typedef __attribute__((ext_vector_type(8)))  __bf16 v8bf;
typedef __attribute__((ext_vector_type(8)))  float  v8f;
typedef __attribute__((ext_vector_type(4)))  float  v4f;

#define DEVFN __device__ __forceinline__

// alignment sample a=(m*4+c4) -> trunk sample c4*7 + t  (nbr t = m<3?m:m+1, ref t=3)
DEVFN int smap(int n, int mode) {
  if (mode == 1) { int m = n >> 2, c = n & 3; int t = (m < 3) ? m : m + 1; return c * 7 + t; }
  if (mode == 2) { int c = n & 3; return c * 7 + 3; }
  return n;
}

DEVFN float actf(float v, int act) {
  if (act == 1) return v >= 0.f ? v : 0.1f * v;   // lrelu(0.1)
  if (act == 2) return fmaxf(v, 0.f);             // relu
  return v;
}

DEVFN int iclamp(int v, int lo, int hi) { return v < lo ? lo : (v > hi ? hi : v); }

DEVFN v16bf cat8(v8bf a, v8bf b) {
  return __builtin_shufflevector(a, b, 0, 1, 2, 3, 4, 5, 6, 7,
                                 8, 9, 10, 11, 12, 13, 14, 15);
}

// ---------------------------------------------------------------------------
// Implicit-GEMM 3x3 conv (pad 1), stride 1|2, two-pointer channel concat,
// optional fp32 residual, activation. One wave = one 16x16 D tile.
// Inputs NHWC bf16 (channel stride 16 on all vector paths, 1 for conv_first).
// ---------------------------------------------------------------------------
struct CP {
  const bf16_t* inA; const bf16_t* inB;
  const bf16_t* wB;                       // fragment-major [nt][ks][lane][e]
  const float*  bias;                     // [cout]
  const float*  resid;                    // optional fp32 NHWC (stride cout)
  float*  outF; bf16_t* outB;             // optional NHWC outputs (stride cout)
  int ns, hin, win, hout, wout, stride;
  int cout, coutPad, act, mapA, mapB;
};

template <int CINSHIFT, int KSTEPS>
__global__ __launch_bounds__(256) void k_conv(CP p) {
  const int lane = threadIdx.x & 31;
  const int wv   = blockIdx.x * (blockDim.x >> 5) + (threadIdx.x >> 5);
  const int M      = p.hout * p.wout;      // always a multiple of 16
  const int mtiles = M >> 4;
  const int ntiles = p.coutPad >> 4;
  if (wv >= p.ns * mtiles * ntiles) return;      // wave-uniform exit
  const int nt = wv % ntiles;
  const int mt = (wv / ntiles) % mtiles;
  const int n  = wv / (ntiles * mtiles);

  const int half = lane >> 4;
  const int lan  = lane & 15;
  const int c8   = half << 3;                    // chunk channel offset
  constexpr int CINA = (CINSHIFT == 0) ? 1 : 16; // channels of tensor A
  const size_t planeIn = (size_t)p.hin * p.win;
  const bf16_t* baseA = p.inA + (size_t)smap(n, p.mapA) * planeIn * CINA;
  const bf16_t* baseB = p.inB ? p.inB + (size_t)smap(n, p.mapB) * planeIn * 16
                              : (const bf16_t*)0;

  const int m0 = (mt << 4) + lan;                // A row (pixel) for this lane
  const int y  = m0 / p.wout;
  const int x  = m0 - y * p.wout;
  const int ys = y * p.stride - 1;
  const int xs = x * p.stride - 1;

  // one 8-wide A chunk = channels [c8, c8+8) of one tap (tap folds after unroll)
  auto loadTap = [&](const bf16_t* base, int tap) -> v8bf {
    v8bf z = {};
    if (tap >= 9) return z;                      // pad tap: folds to constant
    const int ty = tap / 3, tx = tap - 3 * (tap / 3);
    const int iy = ys + ty, ix = xs + tx;
    if ((unsigned)iy >= (unsigned)p.hin || (unsigned)ix >= (unsigned)p.win) return z;
    return *(const v8bf*)(base + ((size_t)iy * p.win + ix) * 16 + c8);  // b128
  };

  const bf16_t* wp = p.wB + (((size_t)nt * KSTEPS * 32 + lane) << 4);
  v8f acc = {};
#pragma unroll
  for (int ks = 0; ks < KSTEPS; ++ks) {
    v16bf a;
    if constexpr (CINSHIFT == 0) {
      // conv_first (cin=1): taps not channel-contiguous -> scalar path (1 step)
#pragma unroll
      for (int e = 0; e < 16; ++e) {
        const int k = ((e < 8) ? e : e + 8) + c8;   // == tap
        float v = 0.f;
        if (k < 9) {
          const int ty = k / 3, tx = k - 3 * (k / 3);
          const int iy = ys + ty, ix = xs + tx;
          if ((unsigned)iy < (unsigned)p.hin && (unsigned)ix < (unsigned)p.win)
            v = (float)baseA[(size_t)iy * p.win + ix];
        }
        a[e] = (__bf16)v;
      }
    } else if constexpr (CINSHIFT == 4) {
      // cin=16: K-block = taps {2ks, 2ks+1}, both chunks from tensor A
      a = cat8(loadTap(baseA, 2 * ks), loadTap(baseA, 2 * ks + 1));
    } else {
      // cin=32 concat: K-block = tap ks; chunk0 from A, chunk1 from B (always)
      a = cat8(loadTap(baseA, ks), loadTap(baseB, ks));
    }
    // B: fragment-major -> two contiguous b128 loads at constant offsets
    const bf16_t* wq = wp + (size_t)ks * 512;
    if (ks + 1 < KSTEPS) __builtin_prefetch(wq + 512, 0, 3);  // global_prefetch
    const v16bf b = cat8(*(const v8bf*)wq, *(const v8bf*)(wq + 8));
    acc = __builtin_amdgcn_wmma_f32_16x16x32_bf16(false, a, false, b, (short)0, acc,
                                                  false, false);
  }
  // Epilogue: D layout M = v + 8*half, N = lan. NHWC: lanes 0..15 store
  // consecutive channels of one pixel (contiguous 32B per half-wave).
  const int oc = (nt << 4) + lan;
  if (oc < p.cout) {
    const float bs = p.bias[oc];
    const size_t nb = (size_t)n * M;
#pragma unroll
    for (int v = 0; v < 8; ++v) {
      const int m = (mt << 4) + v + (half << 3);
      const size_t idx = (nb + m) * p.cout + oc;
      float val = acc[v] + bs;
      if (p.resid) val += p.resid[idx];
      val = actf(val, p.act);
      if (p.outF) p.outF[idx] = val;
      if (p.outB) p.outB[idx] = (__bf16)val;
    }
  }
}

// ---------------------------------------------------------------------------
// Deformable conv (dcn_sep): A built by sigmoid-masked bilinear sampling of
// the fp32 NHWC source; K = tap*16 + c (same packing as conv — einsum K order
// is arbitrary as long as A and B agree). Kpad=160 (5 WMMA steps), cout=16.
// Offsets/mask read once per (pixel,tap) chunk; corners are 2x v4f loads.
// ---------------------------------------------------------------------------
struct DP {
  const float* xsrc;        // NHWC fp32, C=16 (via mapX)
  const float* om;          // NHWC fp32, C=27 (identity map)
  const bf16_t* wB; const float* bias;
  float* outF; bf16_t* outB;
  int ns, h, w, act, mapX;
};

__global__ __launch_bounds__(256) void k_dcn(DP p) {
  const int lane = threadIdx.x & 31;
  const int wv   = blockIdx.x * (blockDim.x >> 5) + (threadIdx.x >> 5);
  const int M = p.h * p.w;                       // multiple of 16
  const int mtiles = M >> 4;
  if (wv >= p.ns * mtiles) return;
  const int mt = wv % mtiles;
  const int n  = wv / mtiles;
  const int half = lane >> 4;
  const int lan  = lane & 15;
  const int c8   = half << 3;

  const float* xbase = p.xsrc + (size_t)smap(n, p.mapX) * M * 16;
  const int m0 = (mt << 4) + lan;
  const float* omp = p.om + ((size_t)n * M + m0) * 27;
  const int y = m0 / p.w;
  const int x = m0 - y * p.w;

  auto sampleChunk = [&](int tap) -> v8bf {
    v8bf z = {};
    if (tap >= 9) return z;                      // pad tap: folds after unroll
    const float dy = omp[2 * tap], dx = omp[2 * tap + 1];
    float mk = omp[18 + tap];
    mk = 1.f / (1.f + __expf(-mk));
    const int ty = tap / 3, tx = tap - 3 * (tap / 3);
    const float py = (float)(y + ty - 1) + dy;
    const float px = (float)(x + tx - 1) + dx;
    const float yf = floorf(py), xf = floorf(px);
    const int y0 = (int)yf, x0 = (int)xf;
    const float fy = py - yf, fx = px - xf;
    const float w00 = (1.f - fy) * (1.f - fx), w01 = (1.f - fy) * fx;
    const float w10 = fy * (1.f - fx),         w11 = fy * fx;
    v4f lo = {}, hi = {};
    auto corner = [&](int yy, int xx, float wt) {
      if ((unsigned)yy < (unsigned)p.h && (unsigned)xx < (unsigned)p.w) {
        const float* cp = xbase + (size_t)(yy * p.w + xx) * 16 + c8;
        lo += (*(const v4f*)cp) * wt;            // 2x b128 per corner
        hi += (*(const v4f*)(cp + 4)) * wt;
      }
    };
    corner(y0, x0, w00); corner(y0, x0 + 1, w01);
    corner(y0 + 1, x0, w10); corner(y0 + 1, x0 + 1, w11);
    lo *= mk; hi *= mk;
    v8bf r;
#pragma unroll
    for (int t = 0; t < 4; ++t) { r[t] = (__bf16)lo[t]; r[4 + t] = (__bf16)hi[t]; }
    return r;
  };

  v8f acc = {};
#pragma unroll
  for (int ks = 0; ks < 5; ++ks) {
    const v16bf a = cat8(sampleChunk(2 * ks), sampleChunk(2 * ks + 1));
    const bf16_t* wq = p.wB + (((size_t)ks * 32 + lane) << 4);
    const v16bf b = cat8(*(const v8bf*)wq, *(const v8bf*)(wq + 8));
    acc = __builtin_amdgcn_wmma_f32_16x16x32_bf16(false, a, false, b, (short)0, acc,
                                                  false, false);
  }
  const float bs = p.bias[lan];
  const size_t nb = (size_t)n * M;
#pragma unroll
  for (int v = 0; v < 8; ++v) {
    const int m = (mt << 4) + v + (half << 3);
    const size_t idx = (nb + m) * 16 + lan;
    const float val = actf(acc[v] + bs, p.act);
    if (p.outF) p.outF[idx] = val;
    if (p.outB) p.outB[idx] = (__bf16)val;
  }
}

// ---------------------------------------------------------------------------
// Weight repack: OIHW fp32 -> fragment-major bf16 [nt][ks][lane][e], value at
// (nt,ks,lane,e) = W[K = ks*32 + (e<8 ? e : e+8) + 8*(lane>>4)][col = nt*16 + (lane&15)]
// with K = tap*cin + c and W[k][o] = w[o][c][tap].  (dcn uses cin=16 too.)
// ---------------------------------------------------------------------------
__global__ void k_packw(const float* w, bf16_t* wB, int cin, int cout, int coutPad,
                        int Ksteps) {
  const int total = (coutPad >> 4) * Ksteps * 512;
  const int i = blockIdx.x * blockDim.x + threadIdx.x;
  if (i >= total) return;
  const int e    = i & 15;
  const int lane = (i >> 4) & 31;
  const int rest = i >> 9;
  const int ks   = rest % Ksteps;
  const int nt   = rest / Ksteps;
  const int k    = ks * 32 + ((e < 8) ? e : e + 8) + ((lane >> 4) << 3);
  const int col  = (nt << 4) + (lane & 15);
  float v = 0.f;
  if (col < cout) {
    const int tap = k / cin, c = k - tap * cin;
    if (tap < 9) v = w[((size_t)col * cin + c) * 9 + tap];
  }
  wB[i] = (__bf16)v;
}

// X (1,7,3,160,160) -> x_in bf16 (28 samples, 1 channel), sample s = c4*7 + t,
// channel 3 = max-min residual.
__global__ void k_pack_input(const float* X, bf16_t* xin) {
  const int i = blockIdx.x * blockDim.x + threadIdx.x;
  const int M = 25600;
  if (i >= 7 * M) return;
  const int t = i / M, p = i - t * M;
  const float r = X[(size_t)(t * 3 + 0) * M + p];
  const float g = X[(size_t)(t * 3 + 1) * M + p];
  const float b = X[(size_t)(t * 3 + 2) * M + p];
  const float mx = fmaxf(r, fmaxf(g, b)), mn = fminf(r, fminf(g, b));
  xin[(size_t)(0 * 7 + t) * M + p] = (__bf16)r;
  xin[(size_t)(1 * 7 + t) * M + p] = (__bf16)g;
  xin[(size_t)(2 * 7 + t) * M + p] = (__bf16)b;
  xin[(size_t)(3 * 7 + t) * M + p] = (__bf16)(mx - mn);
}

// bilinear 2x upsample (half-pixel centers, edge clamp), NHWC fp32 -> NHWC bf16
__global__ void k_up2(const float* in, bf16_t* out, int ns, int h, int w, float scale) {
  const int oh = 2 * h, ow = 2 * w;
  const size_t total = (size_t)ns * oh * ow * 16;
  const size_t i = (size_t)blockIdx.x * blockDim.x + threadIdx.x;
  if (i >= total) return;
  const int c = (int)(i & 15);
  size_t pix = i >> 4;
  const int ox = (int)(pix % ow);
  size_t r = pix / ow;
  const int oy = (int)(r % oh);
  const int s  = (int)(r / oh);
  const float sy = (oy + 0.5f) * 0.5f - 0.5f;
  const float sx = (ox + 0.5f) * 0.5f - 0.5f;
  const float yf = floorf(sy), xf = floorf(sx);
  const int y0 = (int)yf, x0 = (int)xf;
  const float fy = sy - yf, fx = sx - xf;
  const int y0c = iclamp(y0, 0, h - 1), y1c = iclamp(y0 + 1, 0, h - 1);
  const int x0c = iclamp(x0, 0, w - 1), x1c = iclamp(x0 + 1, 0, w - 1);
  const float* b = in + (size_t)s * h * w * 16;
  const float val = b[((size_t)y0c * w + x0c) * 16 + c] * (1.f - fy) * (1.f - fx) +
                    b[((size_t)y0c * w + x1c) * 16 + c] * (1.f - fy) * fx +
                    b[((size_t)y1c * w + x0c) * 16 + c] * fy * (1.f - fx) +
                    b[((size_t)y1c * w + x1c) * 16 + c] * fy * fx;
  out[i] = (__bf16)(val * scale);
}

// aligned NHWC (24, 25600, 16) -> out (1, 64, 6, 160, 160)
__global__ void k_assemble(const float* alignedF, float* out) {
  const size_t M = 25600;
  const size_t i = (size_t)blockIdx.x * blockDim.x + threadIdx.x;
  if (i >= (size_t)24 * M * 16) return;
  const int f = (int)(i & 15);
  const size_t rest = i >> 4;
  const size_t p = rest % M;
  const int a = (int)(rest / M);
  const int m = a >> 2, c4 = a & 3;
  out[((size_t)(c4 * 16 + f) * 6 + m) * M + p] = alignedF[i];
}

// ---------------------------------------------------------------------------
// Host launcher
// ---------------------------------------------------------------------------
extern "C" void kernel_launch(void* const* d_in, const int* in_sizes, int n_in,
                              void* d_out, int out_size, void* d_ws, size_t ws_size,
                              hipStream_t stream) {
  (void)in_sizes; (void)n_in; (void)out_size; (void)ws_size;
  // jax tree_flatten (sorted-key DFS) leaf order:
  auto F = [&](int i) -> const float* { return (const float*)d_in[i]; };
  // 0:X  1/2:L2c1(b,w) 3/4:L2c2 5/6:L3c1 7/8:L3c2
  // 9/10:L1_dcn.conv 11/12:L1_dcn.off 13/14:L1_fc 15/16:L1_oc1 17/18:L1_oc2
  // 19/20:L1_oc3 21/22:L2_dcn.conv 23/24:L2_dcn.off 25/26:L2_fc 27/28:L2_oc1
  // 29/30:L2_oc2 31/32:L2_oc3 33/34:L3_dcn.conv 35/36:L3_dcn.off 37/38:L3_oc1
  // 39/40:L3_oc2 41/42:cas_dcn.conv 43/44:cas_dcn.off 45/46:cas_oc1 47/48:cas_oc2
  // 49/50:conv_first 51:res.b1 52:res.b2 53:res.w1 54:res.w2

  const int Mh = 160 * 160, M2p = 80 * 80, M3p = 40 * 40;
  const int NT = 28, NA = 24;

  size_t off = 0;
  auto A = [&](size_t bytes) -> char* {
    char* r = (char*)d_ws + off;
    off = (off + bytes + 255) & ~(size_t)255;
    return r;
  };

  auto packw = [&](const float* w, int cin, int cout, int coutPad) -> const bf16_t* {
    const int Ksteps = (9 * cin + 31) / 32;
    const int ntot = (coutPad >> 4) * Ksteps * 512;
    bf16_t* wb = (bf16_t*)A((size_t)ntot * sizeof(bf16_t));
    k_packw<<<(ntot + 255) / 256, 256, 0, stream>>>(w, wb, cin, cout, coutPad, Ksteps);
    return wb;
  };

  auto conv = [&](const bf16_t* inA, int mapA, int cinA,
                  const bf16_t* inB, int mapB, int cinB,
                  const bf16_t* wB, const float* bias, const float* resid,
                  float* outF, bf16_t* outB,
                  int ns, int hin, int stride, int cout, int coutPad, int act) {
    CP p{};
    p.inA = inA; p.inB = inB; p.wB = wB; p.bias = bias; p.resid = resid;
    p.outF = outF; p.outB = outB;
    p.ns = ns; p.hin = hin; p.win = hin;
    p.hout = (stride == 2) ? hin / 2 : hin; p.wout = p.hout; p.stride = stride;
    p.cout = cout; p.coutPad = coutPad; p.act = act; p.mapA = mapA; p.mapB = mapB;
    const int cin = cinA + cinB;
    const int M = p.hout * p.wout;
    const long waves = (long)ns * (M >> 4) * (coutPad >> 4);
    const int blocks = (int)((waves + 7) / 8);
    if (cin == 1)       k_conv<0, 1><<<blocks, 256, 0, stream>>>(p);
    else if (cin == 16) k_conv<4, 5><<<blocks, 256, 0, stream>>>(p);
    else                k_conv<5, 9><<<blocks, 256, 0, stream>>>(p);
  };

  auto dcn = [&](const float* xsrc, int mapX, const float* om, const bf16_t* wB,
                 const float* bias, float* outF, bf16_t* outB, int ns, int h, int act) {
    DP p{};
    p.xsrc = xsrc; p.om = om; p.wB = wB; p.bias = bias;
    p.outF = outF; p.outB = outB; p.ns = ns; p.h = h; p.w = h; p.act = act; p.mapX = mapX;
    const long waves = (long)ns * ((h * h) >> 4);
    k_dcn<<<(int)((waves + 7) / 8), 256, 0, stream>>>(p);
  };

  auto up2 = [&](const float* in, bf16_t* out, int ns, int h, float scale) {
    const size_t total = (size_t)ns * (2 * h) * (2 * h) * 16;
    k_up2<<<(int)((total + 255) / 256), 256, 0, stream>>>(in, out, ns, h, h, scale);
  };

  // ---- persistent: packed weights -----------------------------------------
  const bf16_t* wb_first = packw(F(50), 1, 16, 16);
  const bf16_t* wb_r1[5]; const bf16_t* wb_r2[5];
  for (int i = 0; i < 5; ++i) {
    wb_r1[i] = packw(F(53) + (size_t)i * 2304, 16, 16, 16);
    wb_r2[i] = packw(F(54) + (size_t)i * 2304, 16, 16, 16);
  }
  const bf16_t* wb_L2c1 = packw(F(2), 16, 16, 16);
  const bf16_t* wb_L2c2 = packw(F(4), 16, 16, 16);
  const bf16_t* wb_L3c1 = packw(F(6), 16, 16, 16);
  const bf16_t* wb_L3c2 = packw(F(8), 16, 16, 16);
  const bf16_t* wb_L3oc1 = packw(F(38), 32, 16, 16);
  const bf16_t* wb_L3oc2 = packw(F(40), 16, 16, 16);
  const bf16_t* wb_L3off = packw(F(36), 16, 27, 32);
  const bf16_t* wb_L3dcn = packw(F(34), 16, 16, 16);
  const bf16_t* wb_L2oc1 = packw(F(28), 32, 16, 16);
  const bf16_t* wb_L2oc2 = packw(F(30), 32, 16, 16);
  const bf16_t* wb_L2oc3 = packw(F(32), 16, 16, 16);
  const bf16_t* wb_L2off = packw(F(24), 16, 27, 32);
  const bf16_t* wb_L2dcn = packw(F(22), 16, 16, 16);
  const bf16_t* wb_L2fc  = packw(F(26), 32, 16, 16);
  const bf16_t* wb_L1oc1 = packw(F(16), 32, 16, 16);
  const bf16_t* wb_L1oc2 = packw(F(18), 32, 16, 16);
  const bf16_t* wb_L1oc3 = packw(F(20), 16, 16, 16);
  const bf16_t* wb_L1off = packw(F(12), 16, 27, 32);
  const bf16_t* wb_L1dcn = packw(F(10), 16, 16, 16);
  const bf16_t* wb_L1fc  = packw(F(14), 32, 16, 16);
  const bf16_t* wb_cas1   = packw(F(46), 32, 16, 16);
  const bf16_t* wb_cas2   = packw(F(48), 16, 16, 16);
  const bf16_t* wb_casoff = packw(F(44), 16, 27, 32);
  const bf16_t* wb_casdcn = packw(F(42), 16, 16, 16);

  // ---- persistent feature buffers (all NHWC) ------------------------------
  bf16_t* x_in = (bf16_t*)A((size_t)NT * Mh * sizeof(bf16_t));
  float*  L1f  = (float*)A((size_t)NT * Mh * 16 * 4);
  bf16_t* L1b  = (bf16_t*)A((size_t)NT * Mh * 16 * 2);
  float*  L2f  = (float*)A((size_t)NT * M2p * 16 * 4);
  bf16_t* L2b  = (bf16_t*)A((size_t)NT * M2p * 16 * 2);
  float*  L3f  = (float*)A((size_t)NT * M3p * 16 * 4);
  bf16_t* L3b  = (bf16_t*)A((size_t)NT * M3p * 16 * 2);
  // keep-alive across alignment sub-phases
  bf16_t* uo3b = (bf16_t*)A((size_t)NA * M2p * 16 * 2);
  bf16_t* uf3b = (bf16_t*)A((size_t)NA * M2p * 16 * 2);
  bf16_t* uo2b = (bf16_t*)A((size_t)NA * Mh * 16 * 2);
  bf16_t* uf2b = (bf16_t*)A((size_t)NA * Mh * 16 * 2);
  float*  f1f  = (float*)A((size_t)NA * Mh * 16 * 4);
  bf16_t* f1b  = (bf16_t*)A((size_t)NA * Mh * 16 * 2);
  const size_t mark = off;

  // ---- trunk --------------------------------------------------------------
  float*  t0f = (float*)A((size_t)NT * Mh * 16 * 4);
  bf16_t* t0b = (bf16_t*)A((size_t)NT * Mh * 16 * 2);
  bf16_t* t1b = (bf16_t*)A((size_t)NT * Mh * 16 * 2);

  k_pack_input<<<(7 * Mh + 255) / 256, 256, 0, stream>>>(F(0), x_in);
  conv(x_in, 0, 1, nullptr, 0, 0, wb_first, F(49), nullptr, t0f, t0b, NT, 160, 1, 16, 16, 1);
  {
    float* fcF = t0f; bf16_t* fcB = t0b;
    float* faF = L1f; bf16_t* faB = L1b;
    for (int i = 0; i < 5; ++i) {
      conv(fcB, 0, 16, nullptr, 0, 0, wb_r1[i], F(51) + (size_t)i * 16, nullptr,
           nullptr, t1b, NT, 160, 1, 16, 16, 2);
      conv(t1b, 0, 16, nullptr, 0, 0, wb_r2[i], F(52) + (size_t)i * 16, fcF,
           faF, faB, NT, 160, 1, 16, 16, 0);
      { float* tf = fcF; fcF = faF; faF = tf; bf16_t* tb = fcB; fcB = faB; faB = tb; }
    }
    // after 5 swaps the live trunk feature is in L1f/L1b
  }
  conv(L1b, 0, 16, nullptr, 0, 0, wb_L2c1, F(1), nullptr, nullptr, t1b, NT, 160, 2, 16, 16, 1);
  conv(t1b, 0, 16, nullptr, 0, 0, wb_L2c2, F(3), nullptr, L2f, L2b, NT, 80, 1, 16, 16, 1);
  conv(L2b, 0, 16, nullptr, 0, 0, wb_L3c1, F(5), nullptr, nullptr, t1b, NT, 80, 2, 16, 16, 1);
  conv(t1b, 0, 16, nullptr, 0, 0, wb_L3c2, F(7), nullptr, L3f, L3b, NT, 40, 1, 16, 16, 1);
  off = mark;

  // ---- alignment: level 3 -------------------------------------------------
  {
    bf16_t* o3a  = (bf16_t*)A((size_t)NA * M3p * 16 * 2);
    float*  o3f  = (float*)A((size_t)NA * M3p * 16 * 4);
    bf16_t* o3b  = (bf16_t*)A((size_t)NA * M3p * 16 * 2);
    float*  om3f = (float*)A((size_t)NA * M3p * 27 * 4);
    float*  f3f  = (float*)A((size_t)NA * M3p * 16 * 4);
    conv(L3b, 1, 16, L3b, 2, 16, wb_L3oc1, F(37), nullptr, nullptr, o3a, NA, 40, 1, 16, 16, 1);
    conv(o3a, 0, 16, nullptr, 0, 0, wb_L3oc2, F(39), nullptr, o3f, o3b, NA, 40, 1, 16, 16, 1);
    conv(o3b, 0, 16, nullptr, 0, 0, wb_L3off, F(35), nullptr, om3f, nullptr, NA, 40, 1, 27, 32, 0);
    dcn(L3f, 1, om3f, wb_L3dcn, F(33), f3f, nullptr, NA, 40, 1);
    up2(o3f, uo3b, NA, 40, 2.0f);
    up2(f3f, uf3b, NA, 40, 1.0f);
  }
  off = mark;

  // ---- alignment: level 2 -------------------------------------------------
  {
    bf16_t* o2a  = (bf16_t*)A((size_t)NA * M2p * 16 * 2);
    bf16_t* o2b  = (bf16_t*)A((size_t)NA * M2p * 16 * 2);
    float*  o2f  = (float*)A((size_t)NA * M2p * 16 * 4);
    bf16_t* o2bb = (bf16_t*)A((size_t)NA * M2p * 16 * 2);
    float*  om2f = (float*)A((size_t)NA * M2p * 27 * 4);
    bf16_t* f2d  = (bf16_t*)A((size_t)NA * M2p * 16 * 2);
    float*  f2f  = (float*)A((size_t)NA * M2p * 16 * 4);
    conv(L2b, 1, 16, L2b, 2, 16, wb_L2oc1, F(27), nullptr, nullptr, o2a, NA, 80, 1, 16, 16, 1);
    conv(o2a, 0, 16, uo3b, 0, 16, wb_L2oc2, F(29), nullptr, nullptr, o2b, NA, 80, 1, 16, 16, 1);
    conv(o2b, 0, 16, nullptr, 0, 0, wb_L2oc3, F(31), nullptr, o2f, o2bb, NA, 80, 1, 16, 16, 1);
    conv(o2bb, 0, 16, nullptr, 0, 0, wb_L2off, F(23), nullptr, om2f, nullptr, NA, 80, 1, 27, 32, 0);
    dcn(L2f, 1, om2f, wb_L2dcn, F(21), nullptr, f2d, NA, 80, 0);
    conv(f2d, 0, 16, uf3b, 0, 16, wb_L2fc, F(25), nullptr, f2f, nullptr, NA, 80, 1, 16, 16, 1);
    up2(o2f, uo2b, NA, 80, 2.0f);
    up2(f2f, uf2b, NA, 80, 1.0f);
  }
  off = mark;

  // ---- alignment: level 1 + cascading -------------------------------------
  {
    bf16_t* o1a  = (bf16_t*)A((size_t)NA * Mh * 16 * 2);
    bf16_t* o1b  = (bf16_t*)A((size_t)NA * Mh * 16 * 2);
    bf16_t* o1c  = (bf16_t*)A((size_t)NA * Mh * 16 * 2);
    float*  om1f = (float*)A((size_t)NA * Mh * 27 * 4);   // reused for cas offsets
    bf16_t* f1d  = (bf16_t*)A((size_t)NA * Mh * 16 * 2);
    bf16_t* offa = (bf16_t*)A((size_t)NA * Mh * 16 * 2);
    bf16_t* offc = (bf16_t*)A((size_t)NA * Mh * 16 * 2);
    float*  alignedF = (float*)A((size_t)NA * Mh * 16 * 4);
    conv(L1b, 1, 16, L1b, 2, 16, wb_L1oc1, F(15), nullptr, nullptr, o1a, NA, 160, 1, 16, 16, 1);
    conv(o1a, 0, 16, uo2b, 0, 16, wb_L1oc2, F(17), nullptr, nullptr, o1b, NA, 160, 1, 16, 16, 1);
    conv(o1b, 0, 16, nullptr, 0, 0, wb_L1oc3, F(19), nullptr, nullptr, o1c, NA, 160, 1, 16, 16, 1);
    conv(o1c, 0, 16, nullptr, 0, 0, wb_L1off, F(11), nullptr, om1f, nullptr, NA, 160, 1, 27, 32, 0);
    dcn(L1f, 1, om1f, wb_L1dcn, F(9), nullptr, f1d, NA, 160, 0);
    conv(f1d, 0, 16, uf2b, 0, 16, wb_L1fc, F(13), nullptr, f1f, f1b, NA, 160, 1, 16, 16, 0);
    // cascading dcn
    conv(f1b, 0, 16, L1b, 2, 16, wb_cas1, F(45), nullptr, nullptr, offa, NA, 160, 1, 16, 16, 1);
    conv(offa, 0, 16, nullptr, 0, 0, wb_cas2, F(47), nullptr, nullptr, offc, NA, 160, 1, 16, 16, 1);
    conv(offc, 0, 16, nullptr, 0, 0, wb_casoff, F(43), nullptr, om1f, nullptr, NA, 160, 1, 27, 32, 0);
    dcn(f1f, 0, om1f, wb_casdcn, F(41), alignedF, nullptr, NA, 160, 1);
    const size_t tot = (size_t)NA * Mh * 16;
    k_assemble<<<(int)((tot + 255) / 256), 256, 0, stream>>>(alignedF, (float*)d_out);
  }
}
